// E3BaseLineModel_42563125903427
// MI455X (gfx1250) — matrix operand order repaired and checked by reading, not compiled, not run on hardware
//
#include <hip/hip_runtime.h>

typedef _Float16 v16h __attribute__((ext_vector_type(16)));
typedef _Float16 v8h  __attribute__((ext_vector_type(8)));
typedef float    v8f  __attribute__((ext_vector_type(8)));

#define NEDGE   524288
#define RMAX    5.0f
#define WAVES   4
#define THREADS (WAVES * 32)

// fragment bases inside d_ws, in v16h (32-byte) units
#define FRAG_W0 0     // 8 frags  (kt in 0..1, nt in 0..3)
#define FRAG_W1 256   // 16 frags (kt in 0..1, nt in 0..7)
#define FRAG_WE 768   // 40 frags (kt in 0..3, nt in 0..9)

// ---------------------------------------------------------------------------
// Prepack: f32 weights -> f16 WMMA B-fragments in d_ws (32768 f16 = 64 KB).
// B layout: lane half h, elem i<8 holds K=8h+i; elem i>=8 holds K=16+8h+(i-8).
// ---------------------------------------------------------------------------
__global__ __launch_bounds__(256) void e3_prepack(const float* __restrict__ W0,
                                                  const float* __restrict__ W1,
                                                  const float* __restrict__ We,
                                                  _Float16* __restrict__ ws)
{
    int t = blockIdx.x * 256 + threadIdx.x;          // 32768 total
    if (t >= 32768) return;
    float val;
    if (t < 4096) {                                  // W0: Kpad=64 (real 40), N=64
        int f = t >> 9, rem = t & 511, lane = rem >> 4, i = rem & 15;
        int kt = f >> 2, nt = f & 3;
        int K = kt * 32 + (i & 8) * 2 + (lane >> 4) * 8 + (i & 7);
        int n = nt * 16 + (lane & 15);
        val = (K < 40) ? W0[K * 64 + n] * 0.15811388f : 0.0f;      // 1/sqrt(40)
    } else if (t < 12288) {                          // W1: K=64, N=128
        int u = t - 4096;
        int f = u >> 9, rem = u & 511, lane = rem >> 4, i = rem & 15;
        int kt = f >> 3, nt = f & 7;
        int K = kt * 32 + (i & 8) * 2 + (lane >> 4) * 8 + (i & 7);
        int n = nt * 16 + (lane & 15);
        val = W1[K * 128 + n] * 0.20986250f;                        // 1.6789/8
    } else {                                         // W_env: K=128, N=160
        int u = t - 12288;
        int f = u >> 9, rem = u & 511, lane = rem >> 4, i = rem & 15;
        int kt = f / 10, nt = f % 10;
        int K = kt * 32 + (i & 8) * 2 + (lane >> 4) * 8 + (i & 7);
        int n = nt * 16 + (lane & 15);
        val = We[K * 160 + n] * 0.08838835f;                        // 1/sqrt(128)
    }
    ws[t] = (_Float16)val;
}

// ---------------------------------------------------------------------------
// Main kernel: one wave = 16 edges. WMMA chain 40->64->128->160 with f32 acc.
// ---------------------------------------------------------------------------
__global__ __launch_bounds__(THREADS) void e3_main(
    const int*   __restrict__ eidx, const float* __restrict__ esh,
    const float* __restrict__ elen, const float* __restrict__ onehot,
    const float* __restrict__ bw,   const float* __restrict__ tw,
    const _Float16* __restrict__ ws, float* __restrict__ out)
{
    __shared__ __align__(16) _Float16 stage[WAVES][16][128]; // activations (f16)
    __shared__ __align__(16) float    wbuf [WAVES][16][64];  // w tiles (f32)
    __shared__ __align__(16) float    scal [WAVES][16][16];  // per-edge scalars

    const int tid   = threadIdx.x;
    const int wv    = tid >> 5;
    const int l     = tid & 31;
    const int col16 = l & 15;
    const int half  = l >> 4;
    const int eb0   = (blockIdx.x * WAVES + wv) * 16;
    const int edge  = eb0 + col16;          // edge owned during the A/row phase

    const v16h* __restrict__ wsv = (const v16h*)ws;

    // ---- per-edge scalars (both half-waves compute the same edge) ----------
    float r = elen[edge];
    float x = r * (1.0f / RMAX);
    float ebv[8];
#pragma unroll
    for (int j = 0; j < 8; j++) ebv[j] = (2.0f / RMAX) * __sinf(bw[j] * x) / r;
    float x2 = x * x, x3 = x2 * x, x6 = x3 * x3, x7 = x6 * x, x8 = x7 * x;
    float cut = 1.0f - 28.0f * x6 + 48.0f * x7 - 21.0f * x8;
    cut = (x < 1.0f) ? cut : 0.0f;
    float maskf = (cut > 0.0f) ? 1.0f : 0.0f;

    int src = eidx[edge];
    int dst = eidx[NEDGE + edge];

    float s  = esh[edge * 4 + 0];
    float vx = esh[edge * 4 + 1], vy = esh[edge * 4 + 2], vz = esh[edge * 4 + 3];
    const float SQ3 = 1.7320508075688772f;
    float sq  = s * s;
    float dot = (vx * vx + vy * vy + vz * vz) * (1.0f / SQ3);
    if (half == 0) {
        scal[wv][col16][0]  = tw[0] * sq + tw[1] * dot;                 // tp0a
        scal[wv][col16][1]  = tw[4] * sq + tw[5] * dot;                 // tp0b
        scal[wv][col16][2]  = tw[2] * s * vx;                           // tp1a
        scal[wv][col16][3]  = tw[2] * s * vy;
        scal[wv][col16][4]  = tw[2] * s * vz;
        scal[wv][col16][5]  = tw[3] * s * vx;                           // tp1b
        scal[wv][col16][6]  = tw[3] * s * vy;
        scal[wv][col16][7]  = tw[3] * s * vz;
        scal[wv][col16][8]  = tw[6] * SQ3 * vx * vy;                    // tp2
        scal[wv][col16][9]  = tw[6] * SQ3 * vy * vz;
        scal[wv][col16][10] = tw[6] * 0.5f * (2.0f * vz * vz - vx * vx - vy * vy);
        scal[wv][col16][11] = tw[6] * SQ3 * vx * vz;
        scal[wv][col16][12] = tw[6] * 0.5f * SQ3 * (vx * vx - vy * vy);
        scal[wv][col16][13] = cut;
        scal[wv][col16][14] = maskf;
        out[(size_t)NEDGE * 416 + edge] = cut;                          // cut out
    }

    // ---- A fragments for matmul1 (rows = edges; K padded 40 -> 64) ---------
    // ISA 16-bit A layout: lane holds row l%16; l/16 selects K sub-slices.
    v16h a0, a1;
#pragma unroll
    for (int i = 0; i < 8; i++) {
        a0[i]     = (_Float16)onehot[src * 16 + half * 8 + i];   // K=half*8+i
        a0[8 + i] = (_Float16)onehot[dst * 16 + half * 8 + i];   // K=16+half*8+i
        a1[i]     = (half == 0) ? (_Float16)ebv[i] : (_Float16)0.0f; // K=32..39
        a1[8 + i] = (_Float16)0.0f;                               // K=48..63 pad
    }

    // ---- matmul1: h1 = silu(h0 @ W0s), N = 64 (4 tiles, 2 K-steps) ---------
#pragma unroll
    for (int nt = 0; nt < 4; nt++) {
        v8f c = {};
        c = __builtin_amdgcn_wmma_f32_16x16x32_f16(false, a0, false,
                wsv[FRAG_W0 + (0 * 4 + nt) * 32 + l], (short)0, c, false, false);
        c = __builtin_amdgcn_wmma_f32_16x16x32_f16(false, a1, false,
                wsv[FRAG_W0 + (1 * 4 + nt) * 32 + l], (short)0, c, false, false);
#pragma unroll
        for (int rr = 0; rr < 8; rr++) {
            float xv = c[rr];
            float sv = xv / (1.0f + __expf(-xv));                 // SiLU
            stage[wv][rr + 8 * half][nt * 16 + col16] = (_Float16)sv;
        }
    }

    // ---- A fragments for matmul2 from stage (h1, K=64) ---------------------
    v16h a2[2];
#pragma unroll
    for (int ks = 0; ks < 2; ks++) {
        const _Float16* rp = &stage[wv][col16][0];
        v8h lo = *(const v8h*)(rp + ks * 32 + half * 8);
        v8h hi = *(const v8h*)(rp + ks * 32 + 16 + half * 8);
#pragma unroll
        for (int i = 0; i < 8; i++) { a2[ks][i] = lo[i]; a2[ks][8 + i] = hi[i]; }
    }

    // ---- matmul2: latents = mask*cut*(h1 @ W1s), N = 128 (8 tiles) ---------
#pragma unroll
    for (int nt = 0; nt < 8; nt++) {
        v8f c = {};
        c = __builtin_amdgcn_wmma_f32_16x16x32_f16(false, a2[0], false,
                wsv[FRAG_W1 + (0 * 8 + nt) * 32 + l], (short)0, c, false, false);
        c = __builtin_amdgcn_wmma_f32_16x16x32_f16(false, a2[1], false,
                wsv[FRAG_W1 + (1 * 8 + nt) * 32 + l], (short)0, c, false, false);
#pragma unroll
        for (int rr = 0; rr < 8; rr++) {
            int row = rr + 8 * half;
            float lat = scal[wv][row][14] * scal[wv][row][13] * c[rr];
            out[(size_t)(eb0 + row) * 128 + nt * 16 + col16] = lat; // latents out
            stage[wv][row][nt * 16 + col16] = (_Float16)lat;
        }
    }

    // ---- A fragments for matmul3 from stage (latents, K=128) ---------------
    v16h a3[4];
#pragma unroll
    for (int ks = 0; ks < 4; ks++) {
        const _Float16* rp = &stage[wv][col16][0];
        v8h lo = *(const v8h*)(rp + ks * 32 + half * 8);
        v8h hi = *(const v8h*)(rp + ks * 32 + 16 + half * 8);
#pragma unroll
        for (int i = 0; i < 8; i++) { a3[ks][i] = lo[i]; a3[ks][8 + i] = hi[i]; }
    }

    const float inv2 = 0.70710678118654752f;
    float* featp = out + (size_t)NEDGE * 128;

    // ---- matmul3 group 0: w tiles 0..3 -> f0 (feature cols 0..31) ----------
#pragma unroll
    for (int nt = 0; nt < 4; nt++) {
        v8f c = {};
#pragma unroll
        for (int ks = 0; ks < 4; ks++)
            c = __builtin_amdgcn_wmma_f32_16x16x32_f16(false, a3[ks], false,
                    wsv[FRAG_WE + (ks * 10 + nt) * 32 + l], (short)0, c, false, false);
#pragma unroll
        for (int rr = 0; rr < 8; rr++)
            wbuf[wv][rr + 8 * half][nt * 16 + col16] = c[rr];
    }
#pragma unroll 4
    for (int e = 0; e < 16; e++) {
        float val = inv2 * (wbuf[wv][e][l]      * scal[wv][e][0] +
                            wbuf[wv][e][32 + l] * scal[wv][e][1]);
        featp[(size_t)(eb0 + e) * 288 + l] = val;
    }

    // ---- matmul3 group 1: w tiles 4..7 -> f1 (feature cols 32..127) --------
#pragma unroll
    for (int nt = 4; nt < 8; nt++) {
        v8f c = {};
#pragma unroll
        for (int ks = 0; ks < 4; ks++)
            c = __builtin_amdgcn_wmma_f32_16x16x32_f16(false, a3[ks], false,
                    wsv[FRAG_WE + (ks * 10 + nt) * 32 + l], (short)0, c, false, false);
#pragma unroll
        for (int rr = 0; rr < 8; rr++)
            wbuf[wv][rr + 8 * half][(nt - 4) * 16 + col16] = c[rr];
    }
    for (int e = 0; e < 16; e++) {
#pragma unroll
        for (int it = 0; it < 3; it++) {
            int cc = l + 32 * it;             // 0..95 = m*3 + j
            int m = cc / 3, j = cc - 3 * m;
            float val = inv2 * (wbuf[wv][e][m]      * scal[wv][e][2 + j] +
                                wbuf[wv][e][32 + m] * scal[wv][e][5 + j]);
            featp[(size_t)(eb0 + e) * 288 + 32 + cc] = val;
        }
    }

    // ---- matmul3 group 2: w tiles 8..9 -> f2 (feature cols 128..287) -------
#pragma unroll
    for (int nt = 8; nt < 10; nt++) {
        v8f c = {};
#pragma unroll
        for (int ks = 0; ks < 4; ks++)
            c = __builtin_amdgcn_wmma_f32_16x16x32_f16(false, a3[ks], false,
                    wsv[FRAG_WE + (ks * 10 + nt) * 32 + l], (short)0, c, false, false);
#pragma unroll
        for (int rr = 0; rr < 8; rr++)
            wbuf[wv][rr + 8 * half][(nt - 8) * 16 + col16] = c[rr];
    }
    for (int e = 0; e < 16; e++) {
#pragma unroll
        for (int it = 0; it < 5; it++) {
            int cc = l + 32 * it;             // 0..159 = m*5 + j
            int m = cc / 5, j = cc - 5 * m;
            float val = wbuf[wv][e][m] * scal[wv][e][8 + j];
            featp[(size_t)(eb0 + e) * 288 + 128 + cc] = val;
        }
    }
}

// ---------------------------------------------------------------------------
extern "C" void kernel_launch(void* const* d_in, const int* in_sizes, int n_in,
                              void* d_out, int out_size, void* d_ws, size_t ws_size,
                              hipStream_t stream)
{
    (void)in_sizes; (void)n_in; (void)out_size; (void)ws_size;
    const int*   eidx = (const int*)  d_in[0];
    const float* esh  = (const float*)d_in[1];
    const float* elen = (const float*)d_in[2];
    const float* oh   = (const float*)d_in[3];
    const float* bw   = (const float*)d_in[4];
    const float* tw   = (const float*)d_in[5];
    const float* W0   = (const float*)d_in[6];
    const float* W1   = (const float*)d_in[7];
    const float* We   = (const float*)d_in[8];
    _Float16* ws  = (_Float16*)d_ws;   // 64 KB of packed f16 B-fragments
    float*    out = (float*)d_out;

    e3_prepack<<<128, 256, 0, stream>>>(W0, W1, We, ws);
    e3_main<<<NEDGE / (WAVES * 16), THREADS, 0, stream>>>(eidx, esh, elen, oh,
                                                          bw, tw, ws, out);
}
